// SpinalCordCircuit_84559316124235
// MI455X (gfx1250) — compile-verified, matrix operand order checked by reading, not run on hardware
//
#include <hip/hip_runtime.h>
#include <hip/hip_bf16.h>

// ---------------------------------------------------------------------------
// CDNA5 / gfx1250 implementation.
// wave32, WMMA 16x16x32 bf16 (fp32 accumulate). Weights + recurrent state in
// LDS; batch partitioned across workgroups (scan is independent per sample).
// Fragments are loaded as two contiguous 16B runs -> ds_load_b128 pairs.
// Layernorm reductions are wave-level shuffles (16-lane groups), no barriers.
// ---------------------------------------------------------------------------

typedef __attribute__((ext_vector_type(16))) __bf16 v16bf;
typedef __attribute__((ext_vector_type(8)))  __bf16 v8bf;
typedef __attribute__((ext_vector_type(8)))  float  v8f;

#define NN    128      // N (per-half hidden width)
#define NN2   256      // 2N
#define MS    16       // batch rows per workgroup (one WMMA M-tile)
#define NTHR  256      // 8 waves of 32
#define NWAVES 8
#define LN_EPS 1e-5f

// 16-bit A/B fragment (ISA 7.12.2): lane -> row (lane%16), elements are two
// contiguous 8-element K runs: k = e + half*8 (e<8), k = 16 + (e-8) + half*8.
// With ld a multiple of 8 and k0 a multiple of 32, both runs are 16B aligned.
__device__ __forceinline__ v16bf fragA_bf16(const __bf16* p, int ld, int r0, int k0) {
  int lane = threadIdx.x & 31;
  int r    = r0 + (lane & 15);
  int half = lane >> 4;
  const __bf16* base = p + r * ld + k0 + half * 8;
  v8bf lo = *(const v8bf*)(base);        // k = half*8 .. +8
  v8bf hi = *(const v8bf*)(base + 16);   // k = 16+half*8 .. +8
  v16bf f;
#pragma unroll
  for (int e = 0; e < 8; ++e) { f[e] = lo[e]; f[e + 8] = hi[e]; }
  return f;
}

__device__ __forceinline__ v8f wmma_bf16(v16bf a, v16bf b, v8f c) {
  // (neg_a, A, neg_b, B, c_mod, C, reuse_a, reuse_b)
  return __builtin_amdgcn_wmma_f32_16x16x32_bf16(false, a, false, b, (short)0, c,
                                                 false, false);
}

// C/D 16x16 f32 layout: lane gives N, vgpr v -> M = v + (lane/16)*8.
__device__ __forceinline__ void storeC_lds(float* p, int ld, int m0, int n0, v8f c) {
  int lane = threadIdx.x & 31;
  int n    = n0 + (lane & 15);
  int half = lane >> 4;
#pragma unroll
  for (int v = 0; v < 8; ++v)
    p[(m0 + v + half * 8) * ld + n] = c[v];
}

__device__ __forceinline__ float sigmoidf_(float z) { return 1.0f / (1.0f + __expf(-z)); }

// Mean/rstd across the 16 consecutive lanes owning one batch row.
// xor masks 8,4,2,1 stay inside each 16-lane half of the wave. No barriers.
__device__ __forceinline__ void row_stats_w(float s, float ss, float inv_cnt,
                                            float* mean, float* rstd) {
#pragma unroll
  for (int m = 8; m >= 1; m >>= 1) {
    s  += __shfl_xor(s,  m, 32);
    ss += __shfl_xor(ss, m, 32);
  }
  float mu = s * inv_cnt;
  float vv = ss * inv_cnt - mu * mu;
  *mean = mu;
  *rstd = rsqrtf(vv + LN_EPS);
}

// ---------------------------------------------------------------------------
// Grouped 1x1 conv over time:  y[b, g*O+o, t] = sum_c W[g,o,c] * x[b,xoff+g*C+c,t] + bias
// One WG per (t-block of 64, b, g). W ([O][Cin]) and X tile stored t-major
// ([64][Cin]) in LDS so BOTH WMMA operands use the contiguous A-style gather.
// ---------------------------------------------------------------------------
struct GconvArgs {
  const float* x;     // (B, xrows, T)
  const float* W;     // (2, O, Cin)
  const float* bias;  // (2*O)
  float* y;           // (B, 2*O, T)
  int T, Cin, O, xrows, xoff;
};

__global__ void gconv_kernel(GconvArgs a) {
  extern __shared__ char smem[];
  __bf16* Wl = (__bf16*)smem;                                  // [O][Cin]
  __bf16* Xl = (__bf16*)(smem + (size_t)a.O * a.Cin * 2);      // [64][Cin] (t-major)
  const int tid = threadIdx.x;
  const int t0  = blockIdx.x * 64;
  const int b   = blockIdx.y;
  const int g   = blockIdx.z;

  for (int i = tid; i < a.O * a.Cin; i += NTHR)
    Wl[i] = (__bf16)a.W[(size_t)g * a.O * a.Cin + i];
  for (int i = tid; i < a.Cin * 64; i += NTHR) {
    int c = i >> 6, tt = i & 63, t = t0 + tt;       // coalesced over t
    float v = (t < a.T)
        ? a.x[((size_t)b * a.xrows + a.xoff + g * a.Cin + c) * a.T + t] : 0.f;
    Xl[tt * a.Cin + c] = (__bf16)v;                 // transposed store
  }
  __syncthreads();

  const int wave = tid >> 5;
  const int lane = tid & 31;
  const int ntiles = (a.O / 16) * 4;                 // O-tiles x 4 t-tiles
  for (int tile = wave; tile < ntiles; tile += NWAVES) {
    int mo = tile >> 2, nt = tile & 3;
    v8f acc = {};
    for (int kk = 0; kk < a.Cin; kk += 32) {
      v16bf A  = fragA_bf16(Wl, a.Cin, mo * 16, kk);       // A[m,k] = W[m][k]
      v16bf Bm = fragA_bf16(Xl, a.Cin, nt * 16, kk);       // B[k,n] = Xl[n][k]
      acc = wmma_bf16(A, Bm, acc);
    }
    int n = nt * 16 + (lane & 15);
    int t = t0 + n;
    if (t < a.T) {
#pragma unroll
      for (int v = 0; v < 8; ++v) {
        int m = mo * 16 + v + (lane >> 4) * 8;
        a.y[((size_t)b * (2 * a.O) + g * a.O + m) * a.T + t] =
            acc[v] + a.bias[g * a.O + m];
      }
    }
  }
}

// ---------------------------------------------------------------------------
// Recurrent layer (the jax.lax.scan). One WG handles MS=16 batch samples for
// all T steps; weights resident in LDS as bf16; h state fp32 + bf16 mirror.
// ---------------------------------------------------------------------------
struct IntegW {
  const float* h2h_W; const float* h2h_lnw; const float* h2h_lnb;
  const float* x2h_W; const float* x2h_lnw; const float* x2h_lnb;
};
struct LayerArgs {
  const float* src[6];     // pathway base: elem (b,n,t) = src[b*bstride + n*T + t]
  long bstride[6];
  const float* norm_g; const float* norm_b;    // (2P, N)
  const float* af_g;  const float* af_b;       // (P, N)
  const float* ae_g;  const float* ae_b;       // (P, N)
  IntegW flex, ext;
  const float* div_W; const float* div_lnw; const float* div_lnb;   // recip only
  float* out;              // (B, 2N, T)
  int T, P, EI, recip;
};

__device__ __forceinline__ void integ_step(
    const IntegW w, const __bf16* wh, const __bf16* wx,
    float* h, __bf16* hb, float* buf2, float* inb, __bf16* xb,
    float* out, int hoff, int b0, int t, int T) {
  const int tid = threadIdx.x;
  const int b = tid >> 4, sub = tid & 15, n0 = sub << 3, wave = tid >> 5;

  // h2 = h[:, hoff:hoff+N] @ Wh^T  (16 x 256), K=128. A frags shared by tiles.
  {
    v16bf Afr[4];
#pragma unroll
    for (int kk = 0; kk < 4; ++kk)
      Afr[kk] = fragA_bf16(hb, NN2, 0, hoff + kk * 32);
#pragma unroll
    for (int ti2 = 0; ti2 < 2; ++ti2) {
      int ti = wave + ti2 * NWAVES;
      v8f acc = {};
#pragma unroll
      for (int kk = 0; kk < 4; ++kk)
        acc = wmma_bf16(Afr[kk], fragA_bf16(wh, NN, ti * 16, kk * 32), acc);
      storeC_lds(buf2, NN2, 0, ti * 16, acc);
    }
  }
  __syncthreads();

  // LayerNorm(h2) over 256; x = relu(in + h_i); keep x, h_g in regs.
  float xr[8], hg[8];
  {
    float s = 0.f, ss = 0.f;
    int c0 = sub << 4;
#pragma unroll
    for (int j = 0; j < 16; ++j) {
      float v = buf2[b * NN2 + c0 + j];
      s += v; ss += v * v;
    }
    float m, rstd;
    row_stats_w(s, ss, 1.0f / NN2, &m, &rstd);
#pragma unroll
    for (int j = 0; j < 8; ++j) {
      int n = n0 + j;
      float hi = (buf2[b * NN2 + n]      - m) * rstd * w.h2h_lnw[n]      + w.h2h_lnb[n];
      hg[j]    = (buf2[b * NN2 + NN + n] - m) * rstd * w.h2h_lnw[NN + n] + w.h2h_lnb[NN + n];
      float x  = fmaxf(inb[b * NN + n] + hi, 0.f);
      xr[j] = x;
      xb[b * NN + n] = (__bf16)x;
    }
  }
  __syncthreads();   // xb ready; all buf2 reads done

  // gpre = x @ Wx^T (16 x 128), K=128; 8 tiles -> one per wave.
  {
    v8f acc = {};
#pragma unroll
    for (int kk = 0; kk < 4; ++kk) {
      v16bf A  = fragA_bf16(xb, NN, 0, kk * 32);
      v16bf Bm = fragA_bf16(wx, NN, wave * 16, kk * 32);
      acc = wmma_bf16(A, Bm, acc);
    }
    storeC_lds(buf2, NN2, 0, wave * 16, acc);
  }
  __syncthreads();

  // LN(gpre) over 128; g = sigmoid(. + h_g); h' = (1-g)h + g x; emit output.
  {
    float gv[8]; float s = 0.f, ss = 0.f;
#pragma unroll
    for (int j = 0; j < 8; ++j) {
      gv[j] = buf2[b * NN2 + n0 + j];
      s += gv[j]; ss += gv[j] * gv[j];
    }
    float m2, rstd2;
    row_stats_w(s, ss, 1.0f / NN, &m2, &rstd2);
#pragma unroll
    for (int j = 0; j < 8; ++j) {
      int n   = n0 + j;
      float gn = (gv[j] - m2) * rstd2 * w.x2h_lnw[n] + w.x2h_lnb[n];
      float g  = sigmoidf_(gn + hg[j]);
      float ho = h[b * NN2 + hoff + n];
      float hn = (1.f - g) * ho + g * xr[j];
      h[b * NN2 + hoff + n]  = hn;
      hb[b * NN2 + hoff + n] = (__bf16)hn;
      out[((size_t)(b0 + b) * NN2 + hoff + n) * (size_t)T + t] = hn;
    }
  }
  __syncthreads();   // h/hb updates visible before next fragment use; buf2 free
}

__global__ void layer_kernel(LayerArgs a) {
  extern __shared__ char smem[];
  size_t off = 0;
  __bf16* wfh = (__bf16*)(smem + off); off += (size_t)NN2 * NN * 2;  // flex h2h [256][128]
  __bf16* weh = (__bf16*)(smem + off); off += (size_t)NN2 * NN * 2;  // ext  h2h
  __bf16* wfx = (__bf16*)(smem + off); off += (size_t)NN * NN * 2;   // flex x2h [128][128]
  __bf16* wex = (__bf16*)(smem + off); off += (size_t)NN * NN * 2;   // ext  x2h
  float*  h    = (float*)(smem + off); off += (size_t)MS * NN2 * 4;  // state, fp32
  __bf16* hb   = (__bf16*)(smem + off); off += (size_t)MS * NN2 * 2; // state, bf16 mirror
  float*  buf2 = (float*)(smem + off); off += (size_t)MS * NN2 * 4;  // matmul scratch
  float*  infb = (float*)(smem + off); off += (size_t)MS * NN * 4;
  float*  ineb = (float*)(smem + off); off += (size_t)MS * NN * 4;
  __bf16* xb   = (__bf16*)(smem + off); off += (size_t)MS * NN * 2;
  __bf16* wdiv = (__bf16*)(smem + off);                              // recip only [128][256]

  const int tid = threadIdx.x;
  for (int i = tid; i < NN2 * NN; i += NTHR) {
    wfh[i] = (__bf16)a.flex.h2h_W[i];
    weh[i] = (__bf16)a.ext.h2h_W[i];
  }
  for (int i = tid; i < NN * NN; i += NTHR) {
    wfx[i] = (__bf16)a.flex.x2h_W[i];
    wex[i] = (__bf16)a.ext.x2h_W[i];
  }
  if (a.recip)
    for (int i = tid; i < NN * NN2; i += NTHR) wdiv[i] = (__bf16)a.div_W[i];
  for (int i = tid; i < MS * NN2; i += NTHR) { h[i] = 0.f; hb[i] = (__bf16)0.f; }
  __syncthreads();

  const int b = tid >> 4, sub = tid & 15, n0 = sub << 3, wave = tid >> 5;
  const int b0 = blockIdx.x * MS;

  for (int t = 0; t < a.T; ++t) {
    // ---- gather pathway inputs, per-(b,p) layernorm, gate by h, accumulate.
    float aPF[8], aNF[8], aPE[8], aNE[8];
#pragma unroll
    for (int j = 0; j < 8; ++j) { aPF[j] = aNF[j] = aPE[j] = aNE[j] = 0.f; }

    for (int p = 0; p < 2 * a.P; ++p) {
      const int is_e = (p >= a.P) ? 1 : 0;
      const int pp   = is_e ? p - a.P : p;
      const float* src = a.src[p];
      const size_t base = (size_t)(b0 + b) * (size_t)a.bstride[p] + (size_t)t;
      float xv[8]; float s = 0.f, ss = 0.f;
#pragma unroll
      for (int j = 0; j < 8; ++j) {
        float v = src[base + (size_t)(n0 + j) * a.T];
        xv[j] = v; s += v; ss += v * v;
      }
      if (t + 1 < a.T)
        __builtin_prefetch(&src[base + 1 + (size_t)n0 * a.T], 0, 1);  // global_prefetch_b8
      float m, rstd;
      row_stats_w(s, ss, 1.0f / NN, &m, &rstd);
      const float* gg = is_e ? a.ae_g : a.af_g;
      const float* gb = is_e ? a.ae_b : a.af_b;
#pragma unroll
      for (int j = 0; j < 8; ++j) {
        int n = n0 + j;
        float xn = (xv[j] - m) * rstd * a.norm_g[p * NN + n] + a.norm_b[p * NN + n];
        float hv = h[b * NN2 + (is_e ? NN : 0) + n];
        float g  = sigmoidf_(gg[pp * NN + n] * hv + gb[pp * NN + n]);
        float c  = g * xn;
        int neg  = (a.EI && pp == a.P - 1);   // EI_MN = [T,T,F]: last pathway inhibitory
        if (is_e) { if (neg) aNE[j] += c; else aPE[j] += c; }
        else      { if (neg) aNF[j] += c; else aPF[j] += c; }
      }
    }
#pragma unroll
    for (int j = 0; j < 8; ++j) {
      infb[b * NN + n0 + j] = fmaxf(aPF[j], 0.f) - fmaxf(aNF[j], 0.f);
      ineb[b * NN + n0 + j] = fmaxf(aPE[j], 0.f) - fmaxf(aNE[j], 0.f);
    }
    // infb/ineb are produced and consumed by the same (b,n) thread: no barrier.

    // ---- reciprocal divisive gate (Iai only): sigmoid(LN(h @ div_W^T)).
    if (a.recip) {
      v8f acc = {};
#pragma unroll
      for (int kk = 0; kk < 8; ++kk) {
        v16bf A  = fragA_bf16(hb, NN2, 0, kk * 32);
        v16bf Bm = fragA_bf16(wdiv, NN2, wave * 16, kk * 32);
        acc = wmma_bf16(A, Bm, acc);
      }
      storeC_lds(buf2, NN2, 0, wave * 16, acc);
      __syncthreads();
      float gv[8]; float s = 0.f, ss = 0.f;
#pragma unroll
      for (int j = 0; j < 8; ++j) {
        gv[j] = buf2[b * NN2 + n0 + j];
        s += gv[j]; ss += gv[j] * gv[j];
      }
      float m, rstd;
      row_stats_w(s, ss, 1.0f / NN, &m, &rstd);
#pragma unroll
      for (int j = 0; j < 8; ++j) {
        int n = n0 + j;
        float gate = sigmoidf_((gv[j] - m) * rstd * a.div_lnw[n] + a.div_lnb[n]);
        infb[b * NN + n] *= gate;
        ineb[b * NN + n] *= (1.f - gate);
      }
      __syncthreads();   // buf2 reads done before flex integ overwrites it
    }

    // ---- integrators (flex then ext; disjoint halves of h).
    integ_step(a.flex, wfh, wfx, h, hb, buf2, infb, xb, a.out, 0,  b0, t, a.T);
    integ_step(a.ext,  weh, wex, h, hb, buf2, ineb, xb, a.out, NN, b0, t, a.T);
  }
}

// ---------------------------------------------------------------------------
// Host-side orchestration.
// Input order assumes jax-pytree (sorted-key) flattening of setup_inputs().
// ---------------------------------------------------------------------------
static inline IntegW make_integ(void* const* d_in, int i) {
  // nested integ dict sorted: h2h_W, h2h_lnb, h2h_lnw, x2h_W, x2h_lnb, x2h_lnw
  IntegW w;
  w.h2h_W  = (const float*)d_in[i + 0];
  w.h2h_lnb= (const float*)d_in[i + 1];
  w.h2h_lnw= (const float*)d_in[i + 2];
  w.x2h_W  = (const float*)d_in[i + 3];
  w.x2h_lnb= (const float*)d_in[i + 4];
  w.x2h_lnw= (const float*)d_in[i + 5];
  return w;
}

extern "C" void kernel_launch(void* const* d_in, const int* in_sizes, int n_in,
                              void* d_out, int out_size, void* d_ws, size_t ws_size,
                              hipStream_t stream) {
  (void)in_sizes; (void)n_in; (void)out_size; (void)ws_size;
  const int B = 128, T = 1000;

  // --- flattened input indices (sorted nested-dict order) ---
  const float* aff   = (const float*)d_in[0];
  const float* II_W  = (const float*)d_in[1];
  const float* II_b  = (const float*)d_in[2];
  const float* Ia_W  = (const float*)d_in[3];
  const float* Ia_b  = (const float*)d_in[4];
  // Iai: 5 ae_b, 6 ae_g, 7 af_b, 8 af_g, 9 div_W, 10 div_lnb, 11 div_lnw,
  //      12..17 ext, 18..23 flex, 24 norm_b, 25 norm_g
  const float* Iai2mn_W = (const float*)d_in[26];
  const float* Iai2mn_b = (const float*)d_in[27];
  // ex: 28 ae_b, 29 ae_g, 30 af_b, 31 af_g, 32..37 ext, 38..43 flex, 44 norm_b, 45 norm_g
  const float* ex2mn_W  = (const float*)d_in[46];
  const float* ex2mn_b  = (const float*)d_in[47];
  // mn: 48 ae_b, 49 ae_g, 50 af_b, 51 af_g, 52..57 ext, 58..63 flex, 64 norm_b, 65 norm_g

  // --- workspace layout (fp32 elements) ---
  float* ws     = (float*)d_ws;
  float* Ia2    = ws;                                   // (B, 512, T)
  float* II2    = Ia2    + (size_t)B * 512 * T;         // (B, 512, T)
  float* exs    = II2    + (size_t)B * 512 * T;         // (B, 256, T)
  float* ex2mn  = exs    + (size_t)B * 256 * T;         // (B, 256, T)
  float* Iais   = ex2mn  + (size_t)B * 256 * T;         // (B, 256, T)
  float* Iai2mn = Iais   + (size_t)B * 256 * T;         // (B, 256, T)

  const size_t LDS_BASE =
      2 * ((size_t)NN2 * NN * 2) + 2 * ((size_t)NN * NN * 2) +   // weights
      (size_t)MS * NN2 * 4 + (size_t)MS * NN2 * 2 +              // h + hb
      (size_t)MS * NN2 * 4 +                                     // buf2
      2 * ((size_t)MS * NN * 4) + (size_t)MS * NN * 2;           // infb/ineb/xb = 258048
  const size_t LDS_RECIP = LDS_BASE + (size_t)NN * NN2 * 2;      // 323584 < 320KB LDS

  dim3 gblk(NTHR);
  dim3 ggrid((T + 63) / 64, B, 2);

  // 1) gconv Ia: afferents rows [0,128) -> Ia2
  { GconvArgs g{aff, Ia_W, Ia_b, Ia2, T, 64, 256, 256, 0};
    gconv_kernel<<<ggrid, gblk, (size_t)256 * 64 * 2 + 64 * 64 * 2, stream>>>(g); }
  // 2) gconv II: afferents rows [128,256) -> II2
  { GconvArgs g{aff, II_W, II_b, II2, T, 64, 256, 256, 128};
    gconv_kernel<<<ggrid, gblk, (size_t)256 * 64 * 2 + 64 * 64 * 2, stream>>>(g); }

  // 3) layer ex (P=1): paths [II2ex_f (II2 r0), II2ex_e (II2 r256)] -> exs
  { LayerArgs L{};
    L.src[0] = II2 + (size_t)0   * T; L.bstride[0] = (long)512 * T;
    L.src[1] = II2 + (size_t)256 * T; L.bstride[1] = (long)512 * T;
    L.norm_g = (const float*)d_in[45]; L.norm_b = (const float*)d_in[44];
    L.af_g = (const float*)d_in[31]; L.af_b = (const float*)d_in[30];
    L.ae_g = (const float*)d_in[29]; L.ae_b = (const float*)d_in[28];
    L.flex = make_integ(d_in, 38); L.ext = make_integ(d_in, 32);
    L.div_W = nullptr; L.div_lnw = nullptr; L.div_lnb = nullptr;
    L.out = exs; L.T = T; L.P = 1; L.EI = 0; L.recip = 0;
    layer_kernel<<<dim3(B / MS), gblk, LDS_BASE, stream>>>(L); }

  // 4) gconv ex2mn: exs -> ex2mn
  { GconvArgs g{exs, ex2mn_W, ex2mn_b, ex2mn, T, 128, 128, 256, 0};
    gconv_kernel<<<ggrid, gblk, (size_t)128 * 128 * 2 + 128 * 64 * 2, stream>>>(g); }

  // 5) layer Iai (P=2, recip): [Ia2Iai_f r128, II2Iai_f r128, Ia2Iai_e r384, II2Iai_e r384]
  { LayerArgs L{};
    L.src[0] = Ia2 + (size_t)128 * T; L.bstride[0] = (long)512 * T;
    L.src[1] = II2 + (size_t)128 * T; L.bstride[1] = (long)512 * T;
    L.src[2] = Ia2 + (size_t)384 * T; L.bstride[2] = (long)512 * T;
    L.src[3] = II2 + (size_t)384 * T; L.bstride[3] = (long)512 * T;
    L.norm_g = (const float*)d_in[25]; L.norm_b = (const float*)d_in[24];
    L.af_g = (const float*)d_in[8]; L.af_b = (const float*)d_in[7];
    L.ae_g = (const float*)d_in[6]; L.ae_b = (const float*)d_in[5];
    L.flex = make_integ(d_in, 18); L.ext = make_integ(d_in, 12);
    L.div_W = (const float*)d_in[9];
    L.div_lnb = (const float*)d_in[10]; L.div_lnw = (const float*)d_in[11];
    L.out = Iais; L.T = T; L.P = 2; L.EI = 0; L.recip = 1;
    layer_kernel<<<dim3(B / MS), gblk, LDS_RECIP, stream>>>(L); }

  // 6) gconv Iai2mn: Iais -> Iai2mn
  { GconvArgs g{Iais, Iai2mn_W, Iai2mn_b, Iai2mn, T, 128, 128, 256, 0};
    gconv_kernel<<<ggrid, gblk, (size_t)128 * 128 * 2 + 128 * 64 * 2, stream>>>(g); }

  // 7) layer mn (P=3, EI=[T,T,F]):
  //    f: [Ia2mn_f (Ia2 r0), ex2mn_f (r0), Iai2mn_f (r128)]
  //    e: [Ia2mn_e (Ia2 r256), ex2mn_e (r128), Iai2mn_e (r0)]   (note e/f swap in Iai2mn)
  { LayerArgs L{};
    L.src[0] = Ia2    + (size_t)0   * T; L.bstride[0] = (long)512 * T;
    L.src[1] = ex2mn  + (size_t)0   * T; L.bstride[1] = (long)256 * T;
    L.src[2] = Iai2mn + (size_t)128 * T; L.bstride[2] = (long)256 * T;
    L.src[3] = Ia2    + (size_t)256 * T; L.bstride[3] = (long)512 * T;
    L.src[4] = ex2mn  + (size_t)128 * T; L.bstride[4] = (long)256 * T;
    L.src[5] = Iai2mn + (size_t)0   * T; L.bstride[5] = (long)256 * T;
    L.norm_g = (const float*)d_in[65]; L.norm_b = (const float*)d_in[64];
    L.af_g = (const float*)d_in[51]; L.af_b = (const float*)d_in[50];
    L.ae_g = (const float*)d_in[49]; L.ae_b = (const float*)d_in[48];
    L.flex = make_integ(d_in, 58); L.ext = make_integ(d_in, 52);
    L.div_W = nullptr; L.div_lnw = nullptr; L.div_lnb = nullptr;
    L.out = (float*)d_out; L.T = T; L.P = 3; L.EI = 1; L.recip = 0;
    layer_kernel<<<dim3(B / MS), gblk, LDS_BASE, stream>>>(L); }
}
